// GIN2_6098853560655
// MI455X (gfx1250) — matrix operand order repaired and checked by reading, not compiled, not run on hardware
//
#include <hip/hip_runtime.h>
#include <hip/hip_bf16.h>

// ---------------------------------------------------------------------------
// GIN GNN forward on gfx1250 (MI455X).
// GEMMs: V_WMMA_F32_16X16X32_BF16, f32 accumulation, bf16 operands.
//  - B pre-packed into WMMA fragment order (weights reused 3125x) -> lanes
//    fetch 32 contiguous bytes (2x global_load_b128), fully coalesced.
//  - A fetched as 2x b128 per fragment (contiguous 16B runs, 16B aligned).
//  - Each wave computes a 16x32 strip: one A fragment feeds two WMMAs.
// Scatter: per-edge-block coalesced float atomics (x/agg are L2-resident).
// ---------------------------------------------------------------------------

typedef __bf16 bf16_t;
typedef __attribute__((ext_vector_type(16))) __bf16 v16bf;
typedef __attribute__((ext_vector_type(8)))  __bf16 v8bf;
typedef __attribute__((ext_vector_type(8)))  float  v8f;

#define NN     50000
#define EE     800000
#define DIN    128
#define DD     256
#define HH     512
#define NLAYER 3
#define BN_EPS 1e-5f

// ----------------------------- small utility kernels -----------------------

__global__ void zero_f32(float* __restrict__ p, long long n) {
  long long i = (long long)blockIdx.x * blockDim.x + threadIdx.x;
  long long s = (long long)gridDim.x * blockDim.x;
  for (; i < n; i += s) p[i] = 0.0f;
}

__global__ void f32_to_bf16(const float* __restrict__ in, bf16_t* __restrict__ out,
                            long long n) {
  long long i = (long long)blockIdx.x * blockDim.x + threadIdx.x;
  long long s = (long long)gridDim.x * blockDim.x;
  for (; i < n; i += s) out[i] = (bf16_t)in[i];
}

// out_bf = bf16(x + agg)
__global__ void add_to_bf16(const float* __restrict__ x, const float* __restrict__ agg,
                            bf16_t* __restrict__ out, long long n) {
  long long i = (long long)blockIdx.x * blockDim.x + threadIdx.x;
  long long s = (long long)gridDim.x * blockDim.x;
  for (; i < n; i += s) out[i] = (bf16_t)(x[i] + agg[i]);
}

// Pack f32 weight matrix B[K x NC] (row-major) into WMMA B-fragment order:
// P[((tk*(NC/16)+tn)*32 + lane)*16 + e] = B[(tk*32 + (e>>3)*16 + (lane>>4)*8 + (e&7))*NC
//                                           + tn*16 + (lane&15)]
template <int K, int NC>
__global__ void pack_B(const float* __restrict__ B, bf16_t* __restrict__ P) {
  const long long total = (long long)K * NC;
  long long i = (long long)blockIdx.x * blockDim.x + threadIdx.x;
  long long s = (long long)gridDim.x * blockDim.x;
  for (; i < total; i += s) {
    int e    = (int)(i & 15);
    int lane = (int)((i >> 4) & 31);
    long long tile = i >> 9;
    int tn = (int)(tile % (NC / 16));
    int tk = (int)(tile / (NC / 16));
    int k = tk * 32 + (e >> 3) * 16 + (lane >> 4) * 8 + (e & 7);
    int c = tn * 16 + (lane & 15);
    P[i] = (bf16_t)B[(size_t)k * NC + c];
  }
}

// ------------------------------- edge scatter -------------------------------
// 256 consecutive threads (one block) share one edge -> edge-index loads are
// wave-uniform, gather of x[src] is one coalesced 1KB read, atomics coalesced.
__global__ void __launch_bounds__(256)
scatter_add(const float* __restrict__ x, const int* __restrict__ ei,
            float* __restrict__ agg, long long total) {
  long long i = (long long)blockIdx.x * blockDim.x + threadIdx.x;
  long long s = (long long)gridDim.x * blockDim.x;
  for (; i < total; i += s) {
    long long e = i >> 8;            // DD == 256
    int d = (int)(i & 255);
    int src = ei[2 * e + 0];
    int dst = ei[2 * e + 1];
    atomicAdd(&agg[(size_t)dst * DD + d], x[(size_t)src * DD + d]);
  }
}

// ------------------------------- WMMA GEMM ----------------------------------
// C[M x NC] = A(bf16, MxK rowmajor) * Bp(bf16, packed fragments) + bias(+resid)
// Wave computes 16x32 output strip (2 N-tiles, shared A fragment).
// M = 50000 = 3125*16, K/NC multiples of 32 -> no tails, EXEC all-ones at WMMA.
template <int K, int NC>
__global__ void __launch_bounds__(128)
gemm_bf16_wmma(const bf16_t* __restrict__ A, const bf16_t* __restrict__ Bp,
               const float* __restrict__ bias, const float* __restrict__ resid,
               float* __restrict__ C, int M) {
  const int warp  = threadIdx.x >> 5;
  const int lane  = threadIdx.x & 31;
  const int mtile = blockIdx.y * 4 + warp;
  if (mtile * 16 >= M) return;           // whole-wave early out (before WMMA)

  const int m0   = mtile * 16;
  const int n0   = blockIdx.x * 32;      // two 16-wide N tiles
  const int half = lane >> 4;
  const int l16  = lane & 15;
  const int col0 = n0 + l16;
  const int col1 = n0 + 16 + l16;

  v8f acc0, acc1;
#pragma unroll
  for (int r = 0; r < 8; ++r) {
    const size_t crow = (size_t)(m0 + half * 8 + r);
    float v0 = bias[col0], v1 = bias[col1];
    if (resid) { v0 += resid[crow * NC + col0]; v1 += resid[crow * NC + col1]; }
    acc0[r] = v0; acc1[r] = v1;
  }

  // A: row = m0 + (lane&15); lane's data = [half*8 .. +7] and [half*8+16 .. +7]
  const bf16_t* Arow = A + (size_t)(m0 + l16) * K + half * 8;
  // Packed B: 512 bf16 per 32x16 tile; lane's 16 elements are contiguous.
  const int ktileStride = (NC / 16) * 512;
  const bf16_t* Bt = Bp + (size_t)(n0 >> 4) * 512 + lane * 16;

#pragma unroll
  for (int kt = 0; kt < K / 32; ++kt) {
    v8bf alo = *(const v8bf*)(Arow + kt * 32);
    v8bf ahi = *(const v8bf*)(Arow + kt * 32 + 16);
    v16bf a = __builtin_shufflevector(alo, ahi,
        0, 1, 2, 3, 4, 5, 6, 7, 8, 9, 10, 11, 12, 13, 14, 15);

    const bf16_t* bp0 = Bt + (size_t)kt * ktileStride;
    v16bf b0 = __builtin_shufflevector(*(const v8bf*)bp0, *(const v8bf*)(bp0 + 8),
        0, 1, 2, 3, 4, 5, 6, 7, 8, 9, 10, 11, 12, 13, 14, 15);
    v16bf b1 = __builtin_shufflevector(*(const v8bf*)(bp0 + 512), *(const v8bf*)(bp0 + 520),
        0, 1, 2, 3, 4, 5, 6, 7, 8, 9, 10, 11, 12, 13, 14, 15);

    acc0 = __builtin_amdgcn_wmma_f32_16x16x32_bf16(false, a, false, b0,
                                                   (short)0, acc0, false, false);
    acc1 = __builtin_amdgcn_wmma_f32_16x16x32_bf16(false, a, false, b1,
                                                   (short)0, acc1, false, false);
  }

#pragma unroll
  for (int r = 0; r < 8; ++r) {
    const size_t crow = (size_t)(m0 + half * 8 + r);
    C[crow * NC + col0] = acc0[r];
    C[crow * NC + col1] = acc1[r];
  }
}

// --------------------------- BatchNorm column stats -------------------------
template <int COLS>
__global__ void __launch_bounds__(256)
col_stats(const float* __restrict__ X, float* __restrict__ st,
          int rows, int rowsPerBlock) {
  int col = blockIdx.x * blockDim.x + threadIdx.x;
  if (col >= COLS) return;
  int r0 = blockIdx.y * rowsPerBlock;
  int r1 = r0 + rowsPerBlock; if (r1 > rows) r1 = rows;
  float s = 0.0f, s2 = 0.0f;
  for (int r = r0; r < r1; ++r) {
    float v = X[(size_t)r * COLS + col];
    s += v; s2 += v * v;
  }
  atomicAdd(&st[col], s);
  atomicAdd(&st[COLS + col], s2);
}

// BN + ReLU, emit bf16 (operand of the next GEMM)
template <int COLS>
__global__ void bn_relu_to_bf16(const float* __restrict__ X, const float* __restrict__ st,
                                const float* __restrict__ gamma, const float* __restrict__ beta,
                                bf16_t* __restrict__ out, long long n, float invRows) {
  long long i = (long long)blockIdx.x * blockDim.x + threadIdx.x;
  long long s = (long long)gridDim.x * blockDim.x;
  for (; i < n; i += s) {
    int col = (int)(i & (COLS - 1));
    float mean = st[col] * invRows;
    float var  = st[COLS + col] * invRows - mean * mean;
    float inv  = rsqrtf(var + BN_EPS);
    float v = gamma[col] * (X[i] - mean) * inv + beta[col];
    out[i] = (bf16_t)fmaxf(v, 0.0f);
  }
}

// Final BN of a layer, emit f32 (next-layer x, or d_out for the last layer)
template <int COLS>
__global__ void bn_to_f32(const float* __restrict__ X, const float* __restrict__ st,
                          const float* __restrict__ gamma, const float* __restrict__ beta,
                          float* __restrict__ out, long long n, float invRows) {
  long long i = (long long)blockIdx.x * blockDim.x + threadIdx.x;
  long long s = (long long)gridDim.x * blockDim.x;
  for (; i < n; i += s) {
    int col = (int)(i & (COLS - 1));
    float mean = st[col] * invRows;
    float var  = st[COLS + col] * invRows - mean * mean;
    float inv  = rsqrtf(var + BN_EPS);
    out[i] = gamma[col] * (X[i] - mean) * inv + beta[col];
  }
}

// ------------------------------- launcher -----------------------------------

extern "C" void kernel_launch(void* const* d_in, const int* in_sizes, int n_in,
                              void* d_out, int out_size, void* d_ws, size_t ws_size,
                              hipStream_t stream) {
  const float* pre   = (const float*)d_in[0];   // (1,N,128) f32
  const int*   ei    = (const int*)d_in[1];     // (1,E,2) indices
  const float* W0    = (const float*)d_in[2];   // (128,256)
  const float* b0    = (const float*)d_in[3];   // (256,)
  const float* W1    = (const float*)d_in[4];   // (3,256,512)
  const float* b1    = (const float*)d_in[5];   // (3,512)
  const float* g1    = (const float*)d_in[6];   // (3,512)
  const float* beta1 = (const float*)d_in[7];   // (3,512)
  const float* W2    = (const float*)d_in[8];   // (3,512,256)
  const float* b2    = (const float*)d_in[9];   // (3,256)
  const float* gf    = (const float*)d_in[10];  // (3,256)
  const float* betaf = (const float*)d_in[11];  // (3,256)
  float* out = (float*)d_out;

  // ---- carve workspace (all chunk sizes are multiples of 256 bytes) ----
  char* ws = (char*)d_ws;
  float*  x    = (float*)ws;  ws += (size_t)NN * DD * sizeof(float);
  float*  agg  = (float*)ws;  ws += (size_t)NN * DD * sizeof(float);
  float*  tmp  = (float*)ws;  ws += (size_t)NN * DD * sizeof(float);
  float*  h    = (float*)ws;  ws += (size_t)NN * HH * sizeof(float);
  bf16_t* abf  = (bf16_t*)ws; ws += (size_t)NN * HH * sizeof(bf16_t);
  bf16_t* wb0  = (bf16_t*)ws; ws += (size_t)DIN * DD * sizeof(bf16_t);
  bf16_t* wb1  = (bf16_t*)ws; ws += (size_t)NLAYER * DD * HH * sizeof(bf16_t);
  bf16_t* wb2  = (bf16_t*)ws; ws += (size_t)NLAYER * HH * DD * sizeof(bf16_t);
  float*  st   = (float*)ws;  ws += (size_t)2 * HH * sizeof(float);

  const dim3 blk256(256);
  const float invN = 1.0f / (float)NN;

  // ---- one-time: pack weights into WMMA fragment order (bf16) ----
  pack_B<DIN, DD><<<dim3(128), blk256, 0, stream>>>(W0, wb0);
  for (int l = 0; l < NLAYER; ++l) {
    pack_B<DD, HH><<<dim3(512), blk256, 0, stream>>>(
        W1 + (size_t)l * DD * HH, wb1 + (size_t)l * DD * HH);
    pack_B<HH, DD><<<dim3(512), blk256, 0, stream>>>(
        W2 + (size_t)l * HH * DD, wb2 + (size_t)l * HH * DD);
  }
  f32_to_bf16<<<dim3(4096), blk256, 0, stream>>>(pre, abf, (long long)NN * DIN);

  // ---- GEMM0: x = pre @ W0 + b0   [N x 128] @ [128 x 256] ----
  {
    dim3 grid(DD / 32, (NN / 16 + 3) / 4);
    gemm_bf16_wmma<DIN, DD><<<grid, dim3(128), 0, stream>>>(abf, wb0, b0, nullptr, x, NN);
  }

  const long long ND = (long long)NN * DD;
  const long long NH = (long long)NN * HH;
  const long long ED = (long long)EE * DD;

  for (int l = 0; l < NLAYER; ++l) {
    // agg = 0 ; agg[dst] += x[src]
    zero_f32<<<dim3(8192), blk256, 0, stream>>>(agg, ND);
    scatter_add<<<dim3(200000), blk256, 0, stream>>>(x, ei, agg, ED);

    // abf = bf16(x + agg)
    add_to_bf16<<<dim3(8192), blk256, 0, stream>>>(x, agg, abf, ND);

    // h = (x+agg) @ W1[l] + b1[l]   [N x 256] @ [256 x 512]
    {
      dim3 grid(HH / 32, (NN / 16 + 3) / 4);
      gemm_bf16_wmma<DD, HH><<<grid, dim3(128), 0, stream>>>(
          abf, wb1 + (size_t)l * DD * HH, b1 + (size_t)l * HH, nullptr, h, NN);
    }

    // BN(512) + ReLU -> bf16
    zero_f32<<<dim3(4), blk256, 0, stream>>>(st, 2 * HH);
    col_stats<HH><<<dim3(HH / 256, 20), blk256, 0, stream>>>(h, st, NN, 2500);
    bn_relu_to_bf16<HH><<<dim3(8192), blk256, 0, stream>>>(
        h, st, g1 + (size_t)l * HH, beta1 + (size_t)l * HH, abf, NH, invN);

    // tmp = relu_h @ W2[l] + b2[l] + x (residual fused into epilogue)
    {
      dim3 grid(DD / 32, (NN / 16 + 3) / 4);
      gemm_bf16_wmma<HH, DD><<<grid, dim3(128), 0, stream>>>(
          abf, wb2 + (size_t)l * HH * DD, b2 + (size_t)l * DD, x, tmp, NN);
    }

    // BN(256) -> f32 ; last layer writes d_out directly
    zero_f32<<<dim3(2), blk256, 0, stream>>>(st, 2 * DD);
    col_stats<DD><<<dim3(1, 20), blk256, 0, stream>>>(tmp, st, NN, 2500);
    float* dst = (l == NLAYER - 1) ? out : x;
    bn_to_f32<DD><<<dim3(8192), blk256, 0, stream>>>(
        tmp, st, gf + (size_t)l * DD, betaf + (size_t)l * DD, dst, ND, invN);
  }
}
